// MultiHeadAttention2_54820962566690
// MI455X (gfx1250) — matrix operand order repaired and checked by reading, not compiled, not run on hardware
//
#include <hip/hip_runtime.h>
#include <cstdint>

typedef __attribute__((ext_vector_type(16))) _Float16 v16h;
typedef __attribute__((ext_vector_type(8)))  _Float16 v8h;
typedef __attribute__((ext_vector_type(8)))  float    v8f;
typedef __attribute__((ext_vector_type(4)))  unsigned int u32x4;
typedef __attribute__((ext_vector_type(8)))  int          i32x8;
typedef __attribute__((ext_vector_type(4)))  int          i32x4;

#define B_SZ   4
#define S_LEN  2048
#define H_NUM  16
#define HD_DIM 64
#define DMODEL 1024
#define M_ROWS (B_SZ * S_LEN)   /* 8192 */

#define HAS_TDM __has_builtin(__builtin_amdgcn_tensor_load_to_lds)

// ---------------------------------------------------------------------------
// Tensor Data Mover: 2D row-major tile (2-byte elements) global -> LDS.
// D# per CDNA5 ISA ch.8. pad_icode: dwords per pad interval (3->16dw, 4->32dw);
// pad_acode: pad dwords = code+1 (3 -> 4 dwords = 8 halves).
// ---------------------------------------------------------------------------
__device__ __forceinline__ void tdm_load_tile_2d(
    const _Float16* gsrc, const _Float16* ldst,
    unsigned tile_w, unsigned tile_h, unsigned tensor_w,
    unsigned row_stride, int pad_icode, int pad_acode) {
#if HAS_TDM
  unsigned long long ga = (unsigned long long)(uintptr_t)gsrc;
  unsigned la = (unsigned)(uintptr_t)ldst;   // LDS byte offset (addr[31:0])
  u32x4 g0;
  g0.x = 1u;                                         // count=1, user D#
  g0.y = la;                                         // lds_addr
  g0.z = (unsigned)ga;                               // global_addr[31:0]
  g0.w = (unsigned)((ga >> 32) & 0x1FFFFFFull) | (2u << 30);  // hi | type=2

  unsigned d0 = (1u << 16);                          // data_size=1 (2 bytes)
  if (pad_acode >= 0)
    d0 |= (1u << 20) | ((unsigned)pad_icode << 22) | ((unsigned)pad_acode << 25);
  const unsigned td0 = tensor_w;                     // tensor_dim0
  const unsigned td1 = 1u << 20;                     // tensor_dim1 (ample)
  i32x8 g1;
  g1[0] = (int)d0;
  g1[1] = (int)((td0 & 0xFFFFu) << 16);              // tensor_dim0[15:0]
  g1[2] = (int)((td0 >> 16) | ((td1 & 0xFFFFu) << 16));
  g1[3] = (int)((td1 >> 16) | (tile_w << 16));       // tile_dim0
  g1[4] = (int)(tile_h & 0xFFFFu);                   // tile_dim1 (dim2=0)
  g1[5] = (int)row_stride;                           // tensor_dim0_stride lo
  g1[6] = 0;
  g1[7] = 0;
  i32x4 gz;  gz[0] = gz[1] = gz[2] = gz[3] = 0;
  i32x8 gz8;
  for (int i = 0; i < 8; i++) gz8[i] = 0;
  __builtin_amdgcn_tensor_load_to_lds(g0, g1, gz, gz, gz8, 0);
#else
  (void)gsrc; (void)ldst; (void)tile_w; (void)tile_h;
  (void)tensor_w; (void)row_stride; (void)pad_icode; (void)pad_acode;
#endif
}

__device__ __forceinline__ void tdm_wait() {
#if __has_builtin(__builtin_amdgcn_s_wait_tensorcnt)
  __builtin_amdgcn_s_wait_tensorcnt(0);
#else
  asm volatile("s_wait_tensorcnt 0" ::: "memory");
#endif
}

// ---------------------------------------------------------------------------
// fp32 -> fp16 convert (plain) and 1024x1024 transpose-convert (for W^T)
// ---------------------------------------------------------------------------
__global__ void cvt_f32_f16(const float* __restrict__ in,
                            _Float16* __restrict__ out, int n) {
  int i = blockIdx.x * blockDim.x + threadIdx.x;
  if (i < n) out[i] = (_Float16)in[i];
}

__global__ __launch_bounds__(256) void cvt_transpose_f16(
    const float* __restrict__ in, _Float16* __restrict__ out) {
  __shared__ _Float16 t[32][33];
  const int bx = blockIdx.x * 32, by = blockIdx.y * 32;
  const int tx = threadIdx.x & 31, ty = threadIdx.x >> 5;
  for (int r = ty; r < 32; r += 8)
    t[r][tx] = (_Float16)in[(size_t)(by + r) * DMODEL + bx + tx];
  __syncthreads();
  for (int r = ty; r < 32; r += 8)
    out[(size_t)(bx + r) * DMODEL + by + tx] = t[tx][r];
}

// ---------------------------------------------------------------------------
// GEMM: C[8192,1024] = A_f16 @ Wt_f16^T + bias (Wt is pre-transposed [n][k]).
// mode 0: fp32 [M,N].  mode 1: f16 [B,H,S,HD].  mode 2: f16 [B,H,HD,S] (V^T).
// 128 thr = 4 waves; C-tile 128x64; wave owns 32 rows (8 WMMAs / 32-K step).
// All staging via double-buffered TDM (wave0: A-tile, wave1: B-tile).
// ---------------------------------------------------------------------------
__global__ __launch_bounds__(128) void gemm_f16(
    const _Float16* __restrict__ A, const _Float16* __restrict__ Wt,
    const float* __restrict__ bias, void* __restrict__ out,
    int mode, float oscale) {
  __shared__ __align__(16) _Float16 lds_a[2][128][40];  // 32 K + 8 pad
  __shared__ __align__(16) _Float16 lds_b[2][64][40];   // Wt rows [n][k]

  const int tid  = threadIdx.x;
  const int lane = tid & 31;
  const int wave = tid >> 5;
  const int g    = lane >> 4;
  const int lm   = lane & 15;
  const int m0   = blockIdx.y * 128;
  const int n0   = blockIdx.x * 64;

  v8f c[2][4];
  for (int rt = 0; rt < 2; rt++)
    for (int nt = 0; nt < 4; nt++)
      for (int j = 0; j < 8; j++) c[rt][nt][j] = 0.0f;

#if HAS_TDM
  if (wave == 0)
    tdm_load_tile_2d(&A[(size_t)m0 * DMODEL], &lds_a[0][0][0],
                     32, 128, DMODEL, DMODEL, 3, 3);
  else if (wave == 1)
    tdm_load_tile_2d(&Wt[(size_t)n0 * DMODEL], &lds_b[0][0][0],
                     32, 64, DMODEL, DMODEL, 3, 3);
#endif

  for (int k0 = 0; k0 < DMODEL; k0 += 32) {
    const int cur = (k0 >> 5) & 1;
#if HAS_TDM
    if (wave < 2) tdm_wait();
    __syncthreads();
    if (k0 + 32 < DMODEL) {      // prefetch next K-step into other buffer
      if (wave == 0)
        tdm_load_tile_2d(&A[(size_t)m0 * DMODEL + k0 + 32],
                         &lds_a[cur ^ 1][0][0], 32, 128, DMODEL, DMODEL, 3, 3);
      else if (wave == 1)
        tdm_load_tile_2d(&Wt[(size_t)n0 * DMODEL + k0 + 32],
                         &lds_b[cur ^ 1][0][0], 32, 64, DMODEL, DMODEL, 3, 3);
    }
#else
    __syncthreads();
    for (int c8 = tid; c8 < 128 * 4; c8 += 128) {
      int r = c8 >> 2, ch = c8 & 3;
      *(v8h*)&lds_a[cur][r][ch * 8] =
          *(const v8h*)&A[(size_t)(m0 + r) * DMODEL + k0 + ch * 8];
    }
    for (int c8 = tid; c8 < 64 * 4; c8 += 128) {
      int r = c8 >> 2, ch = c8 & 3;
      *(v8h*)&lds_b[cur][r][ch * 8] =
          *(const v8h*)&Wt[(size_t)(n0 + r) * DMODEL + k0 + ch * 8];
    }
    __syncthreads();
#endif

    union Frag { v16h v; v8h h[2]; };
    Frag af[2];
#pragma unroll
    for (int rt = 0; rt < 2; rt++) {
      const int am = wave * 32 + rt * 16 + lm;
      af[rt].h[0] = *(const v8h*)&lds_a[cur][am][g * 8];
      af[rt].h[1] = *(const v8h*)&lds_a[cur][am][16 + g * 8];
    }
#pragma unroll
    for (int nt = 0; nt < 4; nt++) {
      Frag bf;
      bf.h[0] = *(const v8h*)&lds_b[cur][nt * 16 + lm][g * 8];
      bf.h[1] = *(const v8h*)&lds_b[cur][nt * 16 + lm][16 + g * 8];
#pragma unroll
      for (int rt = 0; rt < 2; rt++)
        c[rt][nt] = __builtin_amdgcn_wmma_f32_16x16x32_f16(
            false, af[rt].v, false, bf.v, (short)0, c[rt][nt], false, false);
    }
  }

  // Epilogue: bias (+scale) + store.  C layout: VGPR r -> row r+8g, col lm.
#pragma unroll
  for (int nt = 0; nt < 4; nt++) {
    const int n = n0 + nt * 16 + lm;
    const float bv = bias[n];
#pragma unroll
    for (int rt = 0; rt < 2; rt++) {
#pragma unroll
      for (int r = 0; r < 8; r++) {
        const int m = m0 + wave * 32 + rt * 16 + r + 8 * g;
        const float val = (c[rt][nt][r] + bv) * oscale;
        if (mode == 0) {
          ((float*)out)[(size_t)m * DMODEL + n] = val;
        } else {
          const int b = m >> 11, s = m & (S_LEN - 1);
          const int h = n >> 6, d = n & (HD_DIM - 1);
          size_t idx;
          if (mode == 1)        // [B,H,S,HD]
            idx = ((((size_t)b * H_NUM + h) * S_LEN) + s) * HD_DIM + d;
          else                  // mode 2: [B,H,HD,S]  (V^T)
            idx = ((((size_t)b * H_NUM + h) * HD_DIM) + d) * S_LEN + s;
          ((_Float16*)out)[idx] = (_Float16)val;
        }
      }
    }
  }
}

// ---------------------------------------------------------------------------
// Causal flash attention. One wave per (b,h, 32-query block).
// Q,K: f16 [B,H,S,HD] (Q pre-scaled by 1/sqrt(HD)); Vt: f16 [B,H,HD,S].
// ctx out: f16 [B,S,H*HD].  16 WMMAs per 32-key block; K/V double-buffered TDM.
// ---------------------------------------------------------------------------
__global__ __launch_bounds__(32) void attn_flash(
    const _Float16* __restrict__ Q, const _Float16* __restrict__ K,
    const _Float16* __restrict__ Vt, _Float16* __restrict__ ctx) {
  __shared__ __align__(16) _Float16 lds_q[32][72];      // 64 + 8 pad
  __shared__ __align__(16) _Float16 lds_k[2][32][72];   // K rows [key][d]
  __shared__ __align__(16) _Float16 lds_vT[2][64][40];  // V^T rows [d][key]
  __shared__ __align__(16) _Float16 lds_p[32][40];      // P [q][key]

  const int lane = threadIdx.x;
  const int g    = lane >> 4;
  const int lm   = lane & 15;
  const int bh   = blockIdx.x >> 6;   // / (S/32 = 64)
  const int qb   = blockIdx.x & 63;
  const int q0   = qb * 32;

  const _Float16* Qb  = Q  + (size_t)bh * S_LEN * HD_DIM;
  const _Float16* Kb  = K  + (size_t)bh * S_LEN * HD_DIM;
  const _Float16* Vtb = Vt + (size_t)bh * HD_DIM * S_LEN;

  const int nblk = (q0 + 32 + 31) >> 5;

#if HAS_TDM
  tdm_load_tile_2d(&Qb[(size_t)q0 * HD_DIM], &lds_q[0][0],
                   64, 32, 64, 64, 4, 3);
  tdm_load_tile_2d(Kb, &lds_k[0][0][0], 64, 32, 64, 64, 4, 3);
  tdm_load_tile_2d(Vtb, &lds_vT[0][0][0], 32, 64, S_LEN, S_LEN, 3, 3);
  tdm_wait();
#else
  for (int c8 = lane; c8 < 32 * 8; c8 += 32) {
    int r = c8 >> 3, ch = c8 & 7;
    *(v8h*)&lds_q[r][ch * 8] =
        *(const v8h*)&Qb[(size_t)(q0 + r) * HD_DIM + ch * 8];
  }
#endif
  __syncthreads();

  union Frag { v16h v; v8h h[2]; };
  Frag aq[2][2];  // [row-tile][d-half]
#pragma unroll
  for (int rt = 0; rt < 2; rt++) {
    const int qm = rt * 16 + lm;
    aq[rt][0].h[0] = *(const v8h*)&lds_q[qm][g * 8];
    aq[rt][0].h[1] = *(const v8h*)&lds_q[qm][16 + g * 8];
    aq[rt][1].h[0] = *(const v8h*)&lds_q[qm][32 + g * 8];
    aq[rt][1].h[1] = *(const v8h*)&lds_q[qm][48 + g * 8];
  }

  float mrow[2][8], lrow[2][8];
  v8f acc[2][4];
#pragma unroll
  for (int rt = 0; rt < 2; rt++) {
    for (int r = 0; r < 8; r++) { mrow[rt][r] = -1e30f; lrow[rt][r] = 0.0f; }
    for (int nt = 0; nt < 4; nt++)
      for (int j = 0; j < 8; j++) acc[rt][nt][j] = 0.0f;
  }

  for (int kb = 0; kb < nblk; kb++) {
    const int j0  = kb * 32;
    const int cur = kb & 1;
#if HAS_TDM
    if (kb) { tdm_wait(); }
    __syncthreads();
    if (kb + 1 < nblk) {   // prefetch next key block into other buffer
      tdm_load_tile_2d(&Kb[(size_t)(j0 + 32) * HD_DIM],
                       &lds_k[cur ^ 1][0][0], 64, 32, 64, 64, 4, 3);
      tdm_load_tile_2d(&Vtb[j0 + 32],
                       &lds_vT[cur ^ 1][0][0], 32, 64, S_LEN, S_LEN, 3, 3);
    }
#else
    __syncthreads();
    for (int c8 = lane; c8 < 32 * 8; c8 += 32) {
      int r = c8 >> 3, ch = c8 & 7;
      *(v8h*)&lds_k[cur][r][ch * 8] =
          *(const v8h*)&Kb[(size_t)(j0 + r) * HD_DIM + ch * 8];
    }
    for (int c8 = lane; c8 < 64 * 4; c8 += 32) {
      int r = c8 >> 2, ch = c8 & 3;
      *(v8h*)&lds_vT[cur][r][ch * 8] =
          *(const v8h*)&Vtb[(size_t)r * S_LEN + j0 + ch * 8];
    }
    __syncthreads();
#endif

    // Scores: 2 row-tiles x 2 key-tiles; K-frags loaded once per key-tile
    v8f sc[2][2];
#pragma unroll
    for (int t = 0; t < 2; t++) {
      const int key = t * 16 + lm;
      Frag bk0, bk1;  // B = K^T: element e<8 -> d = g*8+e, e>=8 -> d = 16+g*8+(e-8) (then +32)
      bk0.h[0] = *(const v8h*)&lds_k[cur][key][g * 8];
      bk0.h[1] = *(const v8h*)&lds_k[cur][key][16 + g * 8];
      bk1.h[0] = *(const v8h*)&lds_k[cur][key][32 + g * 8];
      bk1.h[1] = *(const v8h*)&lds_k[cur][key][48 + g * 8];
#pragma unroll
      for (int rt = 0; rt < 2; rt++) {
        v8f s;
        for (int j = 0; j < 8; j++) s[j] = 0.0f;
        s = __builtin_amdgcn_wmma_f32_16x16x32_f16(
            false, aq[rt][0].v, false, bk0.v, (short)0, s, false, false);
        s = __builtin_amdgcn_wmma_f32_16x16x32_f16(
            false, aq[rt][1].v, false, bk1.v, (short)0, s, false, false);
        sc[rt][t] = s;
      }
    }

    const bool need_mask = (j0 + 31 > q0);   // any (q,j) with j > q ?

    // Online softmax per row-tile (rows r + 8g in each 16-lane group)
#pragma unroll
    for (int rt = 0; rt < 2; rt++) {
      float mnew[8], pscale[8];
#pragma unroll
      for (int r = 0; r < 8; r++) {
        const int q = q0 + rt * 16 + r + 8 * g;
        float mx = -1e30f;
#pragma unroll
        for (int t = 0; t < 2; t++) {
          const int j = j0 + t * 16 + lm;
          float v = sc[rt][t][r];
          if (need_mask && j > q) v = -1e30f;   // causal mask
          sc[rt][t][r] = v;
          mx = fmaxf(mx, v);
        }
        for (int off = 1; off < 16; off <<= 1)
          mx = fmaxf(mx, __shfl_xor(mx, off, 16));
        mnew[r]   = fmaxf(mrow[rt][r], mx);
        pscale[r] = __expf(mrow[rt][r] - mnew[r]);
        mrow[rt][r] = mnew[r];
      }
#pragma unroll
      for (int r = 0; r < 8; r++) {
        float sum = 0.0f;
#pragma unroll
        for (int t = 0; t < 2; t++) {
          float p = __expf(sc[rt][t][r] - mnew[r]);
          lds_p[rt * 16 + r + 8 * g][t * 16 + lm] = (_Float16)p;
          sum += p;
        }
        for (int off = 1; off < 16; off <<= 1)
          sum += __shfl_xor(sum, off, 16);
        lrow[rt][r] = lrow[rt][r] * pscale[r] + sum;
#pragma unroll
        for (int nt = 0; nt < 4; nt++) acc[rt][nt][r] *= pscale[r];
      }
    }
    __syncthreads();

    // PV: A = P row-tiles (16x32), B = V dim-subtiles (reused across rt)
    Frag ap[2];
#pragma unroll
    for (int rt = 0; rt < 2; rt++) {
      ap[rt].h[0] = *(const v8h*)&lds_p[rt * 16 + lm][g * 8];
      ap[rt].h[1] = *(const v8h*)&lds_p[rt * 16 + lm][16 + g * 8];
    }
#pragma unroll
    for (int nt = 0; nt < 4; nt++) {
      Frag bv;
      const int d = nt * 16 + lm;
      bv.h[0] = *(const v8h*)&lds_vT[cur][d][g * 8];
      bv.h[1] = *(const v8h*)&lds_vT[cur][d][16 + g * 8];
#pragma unroll
      for (int rt = 0; rt < 2; rt++)
        acc[rt][nt] = __builtin_amdgcn_wmma_f32_16x16x32_f16(
            false, ap[rt].v, false, bv.v, (short)0, acc[rt][nt], false, false);
    }
  }

  // Finalize: ctx[b, q, h, d] = acc / l
  const int b = bh >> 4, h = bh & 15;
#pragma unroll
  for (int nt = 0; nt < 4; nt++) {
    const int d = nt * 16 + lm;
#pragma unroll
    for (int rt = 0; rt < 2; rt++) {
#pragma unroll
      for (int r = 0; r < 8; r++) {
        const int q = q0 + rt * 16 + r + 8 * g;
        float val = acc[rt][nt][r] / lrow[rt][r];
        size_t idx = (((size_t)b * S_LEN + q) * H_NUM + h) * HD_DIM + d;
        ctx[idx] = (_Float16)val;
      }
    }
  }
}

// ---------------------------------------------------------------------------
// Launcher
// ---------------------------------------------------------------------------
extern "C" void kernel_launch(void* const* d_in, const int* in_sizes, int n_in,
                              void* d_out, int out_size, void* d_ws,
                              size_t ws_size, hipStream_t stream) {
  (void)in_sizes; (void)n_in; (void)out_size; (void)ws_size;

  const float* x  = (const float*)d_in[0];
  const float* Wq = (const float*)d_in[1];
  const float* bq = (const float*)d_in[2];
  const float* Wk = (const float*)d_in[3];
  const float* bk = (const float*)d_in[4];
  const float* Wv = (const float*)d_in[5];
  const float* bv = (const float*)d_in[6];
  const float* Wo = (const float*)d_in[7];
  const float* bo = (const float*)d_in[8];

  char* ws = (char*)d_ws;
  const size_t SZ_X = (size_t)M_ROWS * DMODEL * sizeof(_Float16);
  const size_t SZ_W = (size_t)DMODEL * DMODEL * sizeof(_Float16);
  _Float16* xh   = (_Float16*)(ws);
  _Float16* wqt  = (_Float16*)(ws + SZ_X);               // W^T, f16 [n][k]
  _Float16* wkt  = (_Float16*)(ws + SZ_X + SZ_W);
  _Float16* wvt  = (_Float16*)(ws + SZ_X + 2 * SZ_W);
  _Float16* wot  = (_Float16*)(ws + SZ_X + 3 * SZ_W);
  _Float16* qh   = (_Float16*)(ws + SZ_X + 4 * SZ_W);    // [B,H,S,HD]
  _Float16* kh   = (_Float16*)(ws + 2 * SZ_X + 4 * SZ_W);
  _Float16* vth  = (_Float16*)(ws + 3 * SZ_X + 4 * SZ_W);  // [B,H,HD,S]
  _Float16* ctxh = (_Float16*)(ws + 4 * SZ_X + 4 * SZ_W);

  const int nX = M_ROWS * DMODEL;
  cvt_f32_f16<<<(nX + 255) / 256, 256, 0, stream>>>(x, xh, nX);
  dim3 tgrid(DMODEL / 32, DMODEL / 32);   // (32, 32)
  cvt_transpose_f16<<<tgrid, 256, 0, stream>>>(Wq, wqt);
  cvt_transpose_f16<<<tgrid, 256, 0, stream>>>(Wk, wkt);
  cvt_transpose_f16<<<tgrid, 256, 0, stream>>>(Wv, wvt);
  cvt_transpose_f16<<<tgrid, 256, 0, stream>>>(Wo, wot);

  dim3 ggrid(DMODEL / 64, M_ROWS / 128);  // (16, 64)
  const float qscale = 0.125f;            // 1/sqrt(HD)
  gemm_f16<<<ggrid, 128, 0, stream>>>(xh, wqt, bq, (void*)qh, 1, qscale);
  gemm_f16<<<ggrid, 128, 0, stream>>>(xh, wkt, bk, (void*)kh, 1, 1.0f);
  gemm_f16<<<ggrid, 128, 0, stream>>>(xh, wvt, bv, (void*)vth, 2, 1.0f);

  attn_flash<<<B_SZ * H_NUM * (S_LEN / 32), 32, 0, stream>>>(qh, kh, vth, ctxh);

  gemm_f16<<<ggrid, 128, 0, stream>>>(ctxh, wot, bo, d_out, 0, 1.0f);
}
